// cheb_conv_withSAt_240518168585
// MI455X (gfx1250) — compile-verified, matrix operand order checked by reading, not run on hardware
//
#include <hip/hip_runtime.h>
#include <hip/hip_bf16.h>

// ---------------------------------------------------------------------------
// cheb_conv_withSAt for MI455X (gfx1250, wave32, WMMA bf16 16x16x32).
//
// out[b,t,v,o] = relu( sum_{k,u,f} cheb[k,u,v]*sa[b,u,v]*x[b,t,u,f]*theta[k,f,o] )
//
// u' = k*512+u (U'=1536):
//   A'[v,u'] = cheb[k,u,v]*sa[b,u,v]   per-b, t-invariant -> bf16 frags in d_ws
//   Y [u',o] = x[b,t,u,:] @ theta[k]   per-(b,t), bf16 B-frags in LDS
//   out[v,o] = A' x Y                  M=512,N=64,K=1536 bf16 WMMA chain
//
// LDS fragment blocks use a chunk-XOR swizzle and xb uses a 144B row pitch so
// every ds_load_b128 is bank-conflict free (64 banks x 4B).
// ---------------------------------------------------------------------------

#define BB   16
#define TT   64
#define VV   512
#define FF   64
#define OO   64
#define KK   3
#define UP   (KK * VV)     // 1536 folded contraction length
#define NUB  (UP / 32)     // 48 K-blocks of 32 for phase-2
#define NVT  (VV / 16)     // 32 v-tiles
#define NOT  (OO / 16)     // 4 o-tiles

typedef float  v8f   __attribute__((ext_vector_type(8)));
typedef __bf16 v16bf __attribute__((ext_vector_type(16)));

union Frag {            // one wave-matrix operand: 32B per lane
  v16bf v;
  uint4 q[2];
};

__device__ __forceinline__ unsigned pack2_bf16(float lo, float hi) {
  unsigned short l = __builtin_bit_cast(unsigned short, (__bf16)lo);
  unsigned short h = __builtin_bit_cast(unsigned short, (__bf16)hi);
  return (unsigned)l | ((unsigned)h << 16);
}

// Fragment convention (16x32 bf16 A / 32x16 bf16 B, wave32):
// lane L: half h = L/16, row(A)/col(B) = L%16.
// dword r (packs bf16 pair kk,kk+1): kk = h*8+2r (r<4) else 16+h*8+2(r-4).
__device__ __forceinline__ int frag_kk(int h, int r) {
  return (r < 4) ? (h * 8 + 2 * r) : (16 + h * 8 + 2 * (r - 4));
}

// Bank-conflict-free placement of a lane's two 16B chunks inside a 1KB
// LDS fragment block: chunk c -> byte offset L*32 + (c ^ ((L>>3)&1))*16.
__device__ __forceinline__ int chunk_off(int lane, int c) {
  return lane * 32 + ((c ^ ((lane >> 3) & 1)) << 4);
}

// ---------------------------------------------------------------------------
// Prep 1: A'[b][vt][ub] fragments (bf16, swizzled) into d_ws (global: linear
// chunk order, loads are 1KB-contiguous per wave -> perfectly coalesced).
// ---------------------------------------------------------------------------
__global__ __launch_bounds__(256) void prep_a_frags(
    const float* __restrict__ sa, const float* __restrict__ cheb,
    unsigned* __restrict__ afrag) {
  const int bi  = blockIdx.x;            // (b*NVT + vt)*NUB + ub
  const int ub  = bi % NUB;
  const int vt  = (bi / NUB) % NVT;
  const int b   = bi / (NUB * NVT);
  const int tid = threadIdx.x;
  const int lane = tid >> 3, r = tid & 7;
  const int m = lane & 15, h = lane >> 4;
  const int v = vt * 16 + m;
  const int kk  = frag_kk(h, r);
  const int up0 = ub * 32 + kk;          // u' for low bf16; +1 for high
  const int k0 = up0 >> 9, u0 = up0 & 511;
  const int k1 = (up0 + 1) >> 9, u1 = (up0 + 1) & 511;
  const float* sab = sa + (size_t)b * VV * VV;
  const float f0 = cheb[(size_t)k0 * VV * VV + (size_t)u0 * VV + v] * sab[(size_t)u0 * VV + v];
  const float f1 = cheb[(size_t)k1 * VV * VV + (size_t)u1 * VV + v] * sab[(size_t)u1 * VV + v];
  afrag[(size_t)bi * 256 + tid] = pack2_bf16(f0, f1);
}

// ---------------------------------------------------------------------------
// Prep 2: theta (3x64x64 f32) -> bf16 B-fragments, written in the LDS chunk-
// swizzled order so the main kernel's linear ws->LDS copy lands them swizzled.
// ---------------------------------------------------------------------------
__global__ __launch_bounds__(256) void prep_theta_frags(
    const float* __restrict__ theta, unsigned* __restrict__ tfrag) {
  const int g = blockIdx.x * 256 + threadIdx.x;   // 0..6143
  const int d = g & 255, fbk = g >> 8;            // fbk = (k*2+fb)*4 + ot
  const int ot = fbk & 3, fb = (fbk >> 2) & 1, k = fbk >> 3;
  const int lane = d >> 3, r = d & 7;
  const int n = ot * 16 + (lane & 15), h = lane >> 4;
  const int f = fb * 32 + frag_kk(h, r);
  const float lo = theta[(size_t)k * FF * OO + (size_t)f * OO + n];
  const float hi = theta[(size_t)k * FF * OO + (size_t)(f + 1) * OO + n];
  // swizzled destination: chunk = r>>2, dword-in-chunk = r&3
  const int dst = (fbk << 8) + (chunk_off(lane, r >> 3 ? 1 : (r >> 2)) >> 2) + (r & 3);
  tfrag[dst] = pack2_bf16(lo, hi);
}

// ---------------------------------------------------------------------------
// Main kernel: one workgroup (512 thr = 16 waves) per (b,t).
// LDS: xb 72KB (144B row pitch) | theta frags 24KB | Y frags 192KB = 288KB.
// ---------------------------------------------------------------------------
#define XB_PITCH   144                           // bytes per u-row (36 dwords)
#define XB_BYTES   (VV * XB_PITCH)               // 73728
#define TH_OFF     XB_BYTES
#define TH_BYTES   (KK * 2 * NOT * 1024)         // 24576
#define Y_OFF      (TH_OFF + TH_BYTES)           // 98304
#define Y_BYTES    (NUB * NOT * 1024)            // 196608
#define SMEM_TOTAL (Y_OFF + Y_BYTES)             // 294912 <= 320KB WGP budget

__global__ __launch_bounds__(512) void cheb_conv_main(
    const float* __restrict__ x,
    const unsigned* __restrict__ afrag,
    const unsigned* __restrict__ tfrag,
    float* __restrict__ out) {
  extern __shared__ __align__(16) unsigned char smem[];
  const int bt  = blockIdx.x;                    // b*T + t
  const int tid = threadIdx.x;

  // ---- phase 0: x[b,t] f32 -> bf16 into padded-pitch LDS; stage theta ----
  {
    const float4* xs = (const float4*)(x + (size_t)bt * VV * FF);
    uint2* xd = (uint2*)smem;
    #pragma unroll 4
    for (int i = tid; i < VV * FF / 4; i += 512) {   // 16 iters
      float4 p = xs[i];
      const int row = i >> 4, pc = (i & 15) * 2;     // dword col pair
      xd[(row * 36 + pc) >> 1] = make_uint2(pack2_bf16(p.x, p.y),
                                            pack2_bf16(p.z, p.w));
    }
    unsigned* th = (unsigned*)(smem + TH_OFF);
    for (int i = tid; i < TH_BYTES / 4; i += 512) th[i] = tfrag[i];
  }
  __syncthreads();

  const int w = tid >> 5, lane = tid & 31;
  const int m16 = lane & 15, h = lane >> 4;
  const int c0 = chunk_off(lane, 0), c1 = chunk_off(lane, 1);

  // ---- phase 1: Y[k*512+u][o] = xb @ theta_k, u in [w*32, w*32+32) ----
  for (int k = 0; k < KK; ++k) {
    v8f acc[2][NOT];
    #pragma unroll
    for (int ut = 0; ut < 2; ++ut)
      #pragma unroll
      for (int ot = 0; ot < NOT; ++ot)
        #pragma unroll
        for (int i = 0; i < 8; ++i) acc[ut][ot][i] = 0.0f;

    #pragma unroll
    for (int ut = 0; ut < 2; ++ut) {
      #pragma unroll
      for (int fb = 0; fb < 2; ++fb) {
        Frag a;   // rows u = w*32+ut*16+m16; conflict-free via 144B pitch
        const int row = w * 32 + ut * 16 + m16;
        const unsigned char* ap = smem + row * XB_PITCH + fb * 64 + h * 16;
        a.q[0] = *(const uint4*)ap;          // f = fb*32    + h*8 ..+7
        a.q[1] = *(const uint4*)(ap + 32);   // f = fb*32+16 + h*8 ..+7
        #pragma unroll
        for (int ot = 0; ot < NOT; ++ot) {
          Frag bf_;
          const unsigned char* bp =
              smem + TH_OFF + (size_t)(((k * 2 + fb) * NOT) + ot) * 1024;
          bf_.q[0] = *(const uint4*)(bp + c0);
          bf_.q[1] = *(const uint4*)(bp + c1);
          acc[ut][ot] = __builtin_amdgcn_wmma_f32_16x16x32_bf16(
              false, a.v, false, bf_.v, (short)0, acc[ut][ot], false, false);
        }
      }
    }
    // store Y as phase-2 B fragments (swizzled chunks): block ub = k*16 + w
    const int ub = k * 16 + w;
    #pragma unroll
    for (int ot = 0; ot < NOT; ++ot) {
      unsigned dw[8];
      #pragma unroll
      for (int r = 0; r < 4; ++r)
        dw[r] = pack2_bf16(acc[0][ot][2 * r], acc[0][ot][2 * r + 1]);
      #pragma unroll
      for (int r = 0; r < 4; ++r)
        dw[4 + r] = pack2_bf16(acc[1][ot][2 * r], acc[1][ot][2 * r + 1]);
      unsigned char* yb = smem + Y_OFF + (size_t)(ub * NOT + ot) * 1024;
      *(uint4*)(yb + c0) = make_uint4(dw[0], dw[1], dw[2], dw[3]);
      *(uint4*)(yb + c1) = make_uint4(dw[4], dw[5], dw[6], dw[7]);
    }
  }
  __syncthreads();

  // ---- phase 2: out[v,o] = A' x Y, v in [w*32, w*32+32), K = 1536 ----
  v8f oacc[2][NOT];
  #pragma unroll
  for (int vt2 = 0; vt2 < 2; ++vt2)
    #pragma unroll
    for (int ot = 0; ot < NOT; ++ot)
      #pragma unroll
      for (int i = 0; i < 8; ++i) oacc[vt2][ot][i] = 0.0f;

  const int b = bt >> 6;                         // bt / T
  const unsigned* abase0 = afrag + ((size_t)(b * NVT + w * 2) * NUB) * 256;
  const unsigned* abase1 = abase0 + (size_t)NUB * 256;

  for (int ub = 0; ub < NUB; ++ub) {             // 48-step K chain, 8 wmma/step
    Frag a0, a1;
    const uint4* p0 = (const uint4*)(abase0 + (size_t)ub * 256 + lane * 8);
    a0.q[0] = p0[0]; a0.q[1] = p0[1];
    const uint4* p1 = (const uint4*)(abase1 + (size_t)ub * 256 + lane * 8);
    a1.q[0] = p1[0]; a1.q[1] = p1[1];
    #pragma unroll
    for (int ot = 0; ot < NOT; ++ot) {
      Frag yf;
      const unsigned char* yp = smem + Y_OFF + (size_t)(ub * NOT + ot) * 1024;
      yf.q[0] = *(const uint4*)(yp + c0);
      yf.q[1] = *(const uint4*)(yp + c1);
      oacc[0][ot] = __builtin_amdgcn_wmma_f32_16x16x32_bf16(
          false, a0.v, false, yf.v, (short)0, oacc[0][ot], false, false);
      oacc[1][ot] = __builtin_amdgcn_wmma_f32_16x16x32_bf16(
          false, a1.v, false, yf.v, (short)0, oacc[1][ot], false, false);
    }
  }

  // ---- fused ReLU + store (f32): per r the address offset is a constant
  // 256B immediate, halves of the wave write 64B-contiguous runs along o. ----
  float* ob = out + (size_t)bt * VV * OO;
  #pragma unroll
  for (int vt2 = 0; vt2 < 2; ++vt2) {
    const int vrow0 = w * 32 + vt2 * 16 + h * 8;
    #pragma unroll
    for (int ot = 0; ot < NOT; ++ot) {
      const int o = ot * 16 + m16;
      float* obase = ob + (size_t)vrow0 * OO + o;
      #pragma unroll
      for (int r = 0; r < 8; ++r) {
        float val = oacc[vt2][ot][r];
        obase[r * OO] = val > 0.0f ? val : 0.0f;
      }
    }
  }
}

// ---------------------------------------------------------------------------
extern "C" void kernel_launch(void* const* d_in, const int* in_sizes, int n_in,
                              void* d_out, int out_size, void* d_ws, size_t ws_size,
                              hipStream_t stream) {
  const float* x     = (const float*)d_in[0];   // (B,T,V,F) f32
  const float* sa    = (const float*)d_in[1];   // (B,V,V)   f32
  const float* cheb  = (const float*)d_in[2];   // (K,V,V)   f32
  const float* theta = (const float*)d_in[3];   // (K,F,O)   f32
  float* out = (float*)d_out;

  unsigned* afrag = (unsigned*)d_ws;                               // 24 MB bf16 A'
  unsigned* tfrag = afrag + (size_t)BB * NVT * NUB * 256;          // 24 KB bf16 theta

  prep_a_frags<<<BB * NVT * NUB, 256, 0, stream>>>(sa, cheb, afrag);
  prep_theta_frags<<<24, 256, 0, stream>>>(theta, tfrag);
  cheb_conv_main<<<BB * TT, 512, SMEM_TOTAL, stream>>>(x, afrag, tfrag, out);
}
